// StableTransformerLayer_3143916060738
// MI455X (gfx1250) — matrix operand order repaired and checked by reading, not compiled, not run on hardware
//
#include <hip/hip_runtime.h>
#include <math.h>

// ---------------------------------------------------------------------------
// StableTransformerLayer on gfx1250 (MI455X, wave32, WMMA).
// All GEMMs via v_wmma_f32_16x16x32_bf16 (bf16 inputs, f32 accumulate).
// ---------------------------------------------------------------------------

typedef __bf16 bf16_t;
typedef __attribute__((ext_vector_type(16))) __bf16 v16bf;
typedef __attribute__((ext_vector_type(8)))  __bf16 v8bf;
typedef __attribute__((ext_vector_type(8)))  float  v8f;

union ABf { v16bf v; v8bf h[2]; };

__device__ __forceinline__ v8bf ldg8(const bf16_t* p) {
    return *reinterpret_cast<const v8bf*>(p);
}
__device__ __forceinline__ v8f v8f_zero() {
    v8f z;
    #pragma unroll
    for (int i = 0; i < 8; ++i) z[i] = 0.f;
    return z;
}
__device__ __forceinline__ v8f wmma_bf16(const ABf& a, const ABf& b, v8f c) {
    return __builtin_amdgcn_wmma_f32_16x16x32_bf16(
        /*neg_a=*/false, a.v, /*neg_b=*/false, b.v,
        /*c_mod=*/(short)0, c, /*reuse_a=*/false, /*reuse_b=*/false);
}

// Model constants
#define DMODEL 1024
#define DFF    4096
#define SEQL   2048
#define NHEADS 16
#define HDIM   64
#define NTOK   4096   // B * L

// ---------------------------------------------------------------------------
// Weight transpose + fp32 -> bf16 convert:  out[N,K] = (bf16) in[K,N]
// ---------------------------------------------------------------------------
__global__ void stl_transpose_cvt(const float* __restrict__ in,
                                  bf16_t* __restrict__ out, int K, int N) {
    __shared__ float tile[32][33];
    int k0 = blockIdx.y * 32, n0 = blockIdx.x * 32;
    int x = threadIdx.x, y = threadIdx.y;           // blockDim = (32, 8)
    #pragma unroll
    for (int j = 0; j < 32; j += 8)
        tile[y + j][x] = in[(size_t)(k0 + y + j) * N + (n0 + x)];
    __syncthreads();
    #pragma unroll
    for (int j = 0; j < 32; j += 8)
        out[(size_t)(n0 + y + j) * K + (k0 + x)] = (bf16_t)tile[x][y + j];
}

// ---------------------------------------------------------------------------
// LayerNorm over last dim (1024) : fp32 in -> bf16 out.  One block per row.
// ---------------------------------------------------------------------------
__global__ __launch_bounds__(256)
void stl_layernorm_bf16(const float* __restrict__ x, const float* __restrict__ g,
                        const float* __restrict__ bparam, bf16_t* __restrict__ out) {
    __shared__ float red[256];
    int row = blockIdx.x, t = threadIdx.x;
    const float* xr = x + (size_t)row * DMODEL;
    float v[4], s = 0.f, ss = 0.f;
    #pragma unroll
    for (int j = 0; j < 4; ++j) {
        v[j] = xr[t + j * 256];
        s  += v[j];
        ss += v[j] * v[j];
    }
    red[t] = s; __syncthreads();
    for (int o = 128; o > 0; o >>= 1) { if (t < o) red[t] += red[t + o]; __syncthreads(); }
    float mean = red[0] * (1.f / DMODEL); __syncthreads();
    red[t] = ss; __syncthreads();
    for (int o = 128; o > 0; o >>= 1) { if (t < o) red[t] += red[t + o]; __syncthreads(); }
    float var = red[0] * (1.f / DMODEL) - mean * mean;
    float rstd = rsqrtf(var + 1e-5f);
    bf16_t* orow = out + (size_t)row * DMODEL;
    #pragma unroll
    for (int j = 0; j < 4; ++j) {
        int c = t + j * 256;
        orow[c] = (bf16_t)((v[j] - mean) * rstd * g[c] + bparam[c]);
    }
}

// ---------------------------------------------------------------------------
// WMMA GEMM:  C[M,N] = A[M,K](bf16, row-major) * Bt[N,K](bf16, pre-transposed)
//             + bias[N]  (+ GELU) (+ residual, f32 out) | bf16 out | V^T out.
// Block = 256 thr = 8 waves arranged 2(M) x 4(N); block tile 128x128;
// wave tile 64x32 = 4x2 WMMA 16x16 tiles; K step 32.
// Fragments are loaded straight from global: per the CDNA5 VGPR layouts both
// A and B fragments are two contiguous 16B chunks per lane (L2-resident data).
// ---------------------------------------------------------------------------
enum { OUT_BF16 = 0, OUT_VT = 1, OUT_F32RES = 2 };

template <int MODE, bool GELU>
__global__ __launch_bounds__(256)
void stl_gemm_wmma(const bf16_t* __restrict__ A, const bf16_t* __restrict__ Bt,
                   const float* __restrict__ bias, const float* __restrict__ resid,
                   void* __restrict__ out, int M, int N, int K) {
    const int lane = threadIdx.x & 31;
    const int w    = threadIdx.x >> 5;
    const int wm   = w & 1, wn = w >> 1;
    const int mW = blockIdx.y * 128 + wm * 64;
    const int nW = blockIdx.x * 128 + wn * 32;
    const int l16 = lane & 15;
    const int hi  = lane >> 4;       // 0: lanes 0-15, 1: lanes 16-31
    const int aoff = hi * 8;         // A frag K-half offset
    const int boff = hi * 8;         // B frag K offset

    v8f acc[4][2];
    #pragma unroll
    for (int mt = 0; mt < 4; ++mt)
        #pragma unroll
        for (int nt = 0; nt < 2; ++nt) acc[mt][nt] = v8f_zero();

    for (int k = 0; k < K; k += 32) {
        ABf a[4], b[2];
        #pragma unroll
        for (int mt = 0; mt < 4; ++mt) {
            const bf16_t* p = A + (size_t)(mW + mt * 16 + l16) * K + k + aoff;
            a[mt].h[0] = ldg8(p);
            a[mt].h[1] = ldg8(p + 16);
        }
        #pragma unroll
        for (int nt = 0; nt < 2; ++nt) {
            const bf16_t* p = Bt + (size_t)(nW + nt * 16 + l16) * K + k + boff;
            b[nt].h[0] = ldg8(p);
            b[nt].h[1] = ldg8(p + 16);
        }
        #pragma unroll
        for (int mt = 0; mt < 4; ++mt)
            #pragma unroll
            for (int nt = 0; nt < 2; ++nt)
                acc[mt][nt] = wmma_bf16(a[mt], b[nt], acc[mt][nt]);
    }

    const int moff = hi * 8;         // C layout: VGPR r -> row r (+8 for hi half)
    #pragma unroll
    for (int nt = 0; nt < 2; ++nt) {
        const int ncol = nW + nt * 16 + l16;
        const float bv = bias ? bias[ncol] : 0.f;
        #pragma unroll
        for (int mt = 0; mt < 4; ++mt) {
            #pragma unroll
            for (int r = 0; r < 8; ++r) {
                const int mrow = mW + mt * 16 + r + moff;
                float v = acc[mt][nt][r] + bv;
                if (GELU) v = 0.5f * v * (1.f + erff(v * 0.70710678118654752f));
                if (MODE == OUT_BF16) {
                    ((bf16_t*)out)[(size_t)mrow * N + ncol] = (bf16_t)v;
                } else if (MODE == OUT_VT) {
                    // per-head transposed V: Vt[(b*1024 + ncol) * L + pos]
                    int bb = mrow >> 11, pos = mrow & (SEQL - 1);
                    ((bf16_t*)out)[((size_t)(bb * DMODEL + ncol)) * SEQL + pos] = (bf16_t)v;
                } else {
                    size_t idx = (size_t)mrow * N + ncol;
                    ((float*)out)[idx] = v + resid[idx];
                }
            }
        }
    }
}

// ---------------------------------------------------------------------------
// Causal flash attention, one wave per 16-row Q tile.
// Q,K: bf16 [B*L, 1024] row-major; Vt: bf16 [B*H*64, L]; out: bf16 [B*L,1024].
// S = QK^T (2 key sub-tiles x 2 d-chunks WMMA), online softmax, P staged
// through LDS (C-layout -> A-layout), O += P*V (4 WMMA).  scale = 1/8.
// ---------------------------------------------------------------------------
__global__ __launch_bounds__(128)
void stl_attn_fa(const bf16_t* __restrict__ Q, const bf16_t* __restrict__ Kmat,
                 const bf16_t* __restrict__ Vt, bf16_t* __restrict__ O) {
    __shared__ __align__(16) bf16_t pbuf[4][16 * 32];
    const int lane = threadIdx.x & 31;
    const int w    = threadIdx.x >> 5;
    const int bh = blockIdx.y, b = bh >> 4, h = bh & 15;
    const int q0 = blockIdx.x * 64 + w * 16;
    const int l16 = lane & 15;
    const int hi  = lane >> 4;

    // Q fragments (d = 0..31, 32..63)
    ABf qf[2];
    #pragma unroll
    for (int dc = 0; dc < 2; ++dc) {
        const bf16_t* p = Q + (size_t)(b * SEQL + q0 + l16) * DMODEL
                            + h * HDIM + dc * 32 + hi * 8;
        qf[dc].h[0] = ldg8(p);
        qf[dc].h[1] = ldg8(p + 16);
    }

    v8f o[4];
    #pragma unroll
    for (int dt = 0; dt < 4; ++dt) o[dt] = v8f_zero();
    float mr[8], lr[8];
    #pragma unroll
    for (int r = 0; r < 8; ++r) { mr[r] = -1e30f; lr[r] = 0.f; }

    bf16_t* myp = pbuf[w];

    for (int k0 = 0; k0 < q0 + 16; k0 += 32) {
        // ---- S = Q * K^T for 2 key sub-tiles of 16 ----
        v8f s[2];
        s[0] = v8f_zero(); s[1] = v8f_zero();
        #pragma unroll
        for (int nt = 0; nt < 2; ++nt) {
            #pragma unroll
            for (int dc = 0; dc < 2; ++dc) {
                ABf kf;
                const bf16_t* p = Kmat + (size_t)(b * SEQL + k0 + nt * 16 + l16) * DMODEL
                                        + h * HDIM + dc * 32 + hi * 8;
                kf.h[0] = ldg8(p);
                kf.h[1] = ldg8(p + 16);
                s[nt] = wmma_bf16(qf[dc], kf, s[nt]);
            }
        }
        // ---- scale + causal mask ----
        #pragma unroll
        for (int nt = 0; nt < 2; ++nt) {
            const int kcol = k0 + nt * 16 + l16;
            #pragma unroll
            for (int r = 0; r < 8; ++r) {
                const int qglob = q0 + r + hi * 8;
                float v = s[nt][r] * 0.125f;
                s[nt][r] = (kcol > qglob) ? -1e30f : v;
            }
        }
        // ---- online softmax (row stats live across the 16 lanes of each half) ----
        #pragma unroll
        for (int r = 0; r < 8; ++r) {
            float rv = fmaxf(s[0][r], s[1][r]);
            #pragma unroll
            for (int off = 1; off < 16; off <<= 1)
                rv = fmaxf(rv, __shfl_xor(rv, off, 32));
            const float mnew  = fmaxf(mr[r], rv);
            const float alpha = __expf(mr[r] - mnew);
            s[0][r] = __expf(s[0][r] - mnew);
            s[1][r] = __expf(s[1][r] - mnew);
            float rs = s[0][r] + s[1][r];
            #pragma unroll
            for (int off = 1; off < 16; off <<= 1)
                rs += __shfl_xor(rs, off, 32);
            lr[r] = lr[r] * alpha + rs;
            mr[r] = mnew;
            #pragma unroll
            for (int dt = 0; dt < 4; ++dt) o[dt][r] *= alpha;
            // stage P tile to LDS (C layout scatter)
            myp[(r + hi * 8) * 32 +      l16] = (bf16_t)s[0][r];
            myp[(r + hi * 8) * 32 + 16 + l16] = (bf16_t)s[1][r];
        }
        // wave-local DS ordering: all stores visible before the A-layout reads
        asm volatile("s_wait_dscnt 0" ::: "memory");

        ABf pf;
        const bf16_t* pp = myp + l16 * 32 + hi * 8;
        pf.h[0] = *reinterpret_cast<const v8bf*>(pp);
        pf.h[1] = *reinterpret_cast<const v8bf*>(pp + 16);

        // ---- O += P * V ----
        #pragma unroll
        for (int dt = 0; dt < 4; ++dt) {
            ABf vf;
            const bf16_t* vp = Vt + (size_t)(bh * HDIM + dt * 16 + l16) * SEQL
                                   + k0 + hi * 8;
            vf.h[0] = ldg8(vp);
            vf.h[1] = ldg8(vp + 16);
            o[dt] = wmma_bf16(pf, vf, o[dt]);
        }
    }

    // ---- normalize + store ----
    #pragma unroll
    for (int dt = 0; dt < 4; ++dt) {
        #pragma unroll
        for (int r = 0; r < 8; ++r) {
            float val = o[dt][r] / lr[r];
            O[(size_t)(b * SEQL + q0 + r + hi * 8) * DMODEL
              + h * HDIM + dt * 16 + l16] = (bf16_t)val;
        }
    }
}

// ---------------------------------------------------------------------------
// Host-side orchestration.
// ---------------------------------------------------------------------------
extern "C" void kernel_launch(void* const* d_in, const int* in_sizes, int n_in,
                              void* d_out, int out_size, void* d_ws, size_t ws_size,
                              hipStream_t stream) {
    (void)in_sizes; (void)n_in; (void)out_size; (void)ws_size;
    const float* x   = (const float*)d_in[0];
    // d_in[1] = causal mask (implemented analytically)
    const float* Wq  = (const float*)d_in[2];
    const float* bq  = (const float*)d_in[3];
    const float* Wk  = (const float*)d_in[4];
    const float* bk  = (const float*)d_in[5];
    const float* Wv  = (const float*)d_in[6];
    const float* bv  = (const float*)d_in[7];
    const float* Wo  = (const float*)d_in[8];
    const float* bo  = (const float*)d_in[9];
    const float* g1  = (const float*)d_in[10];
    const float* b1  = (const float*)d_in[11];
    const float* g2  = (const float*)d_in[12];
    const float* b2  = (const float*)d_in[13];
    const float* W1  = (const float*)d_in[14];
    const float* bf1 = (const float*)d_in[15];
    const float* W2  = (const float*)d_in[16];
    const float* bf2 = (const float*)d_in[17];

    // Workspace layout (bf16 elements unless noted); total = 80 MB.
    const size_t MEG = 1024u * 1024u;
    bf16_t* wsb = (bf16_t*)d_ws;
    bf16_t* WqT = wsb + 0 * MEG;     // [1024,1024]
    bf16_t* WkT = wsb + 1 * MEG;
    bf16_t* WvT = wsb + 2 * MEG;
    bf16_t* WoT = wsb + 3 * MEG;
    bf16_t* W1T = wsb + 4 * MEG;     // [4096,1024]
    bf16_t* W2T = wsb + 8 * MEG;     // [1024,4096]
    bf16_t* NX  = wsb + 12 * MEG;    // ln out / attn out / ln2 out [4096,1024]
    bf16_t* Qb  = wsb + 16 * MEG;    // [4096,1024]
    bf16_t* Kb  = wsb + 20 * MEG;
    bf16_t* Vt  = wsb + 24 * MEG;    // [B*H*64, 2048]
    bf16_t* Hb  = wsb + 16 * MEG;    // FFN hidden [4096,4096] (reuses Q/K/Vt)
    float*  X1  = (float*)(wsb + 32 * MEG);  // post-attn residual [4096,1024]

    dim3 tb(32, 8);
    // Weight transpose + convert (out[N,K] = in[K,N])
    stl_transpose_cvt<<<dim3(32, 32),  tb, 0, stream>>>(Wq, WqT, DMODEL, DMODEL);
    stl_transpose_cvt<<<dim3(32, 32),  tb, 0, stream>>>(Wk, WkT, DMODEL, DMODEL);
    stl_transpose_cvt<<<dim3(32, 32),  tb, 0, stream>>>(Wv, WvT, DMODEL, DMODEL);
    stl_transpose_cvt<<<dim3(32, 32),  tb, 0, stream>>>(Wo, WoT, DMODEL, DMODEL);
    stl_transpose_cvt<<<dim3(128, 32), tb, 0, stream>>>(W1, W1T, DMODEL, DFF);
    stl_transpose_cvt<<<dim3(32, 128), tb, 0, stream>>>(W2, W2T, DFF, DMODEL);

    // LN1
    stl_layernorm_bf16<<<NTOK, 256, 0, stream>>>(x, g1, b1, NX);

    // Q, K, V projections
    stl_gemm_wmma<OUT_BF16,  false><<<dim3(8, 32), 256, 0, stream>>>(
        NX, WqT, bq, nullptr, Qb, NTOK, DMODEL, DMODEL);
    stl_gemm_wmma<OUT_BF16,  false><<<dim3(8, 32), 256, 0, stream>>>(
        NX, WkT, bk, nullptr, Kb, NTOK, DMODEL, DMODEL);
    stl_gemm_wmma<OUT_VT,    false><<<dim3(8, 32), 256, 0, stream>>>(
        NX, WvT, bv, nullptr, Vt, NTOK, DMODEL, DMODEL);

    // Attention -> NX (bf16)
    stl_attn_fa<<<dim3(SEQL / 64, 2 * NHEADS), 128, 0, stream>>>(Qb, Kb, Vt, NX);

    // Output projection + residual -> X1 (f32)
    stl_gemm_wmma<OUT_F32RES, false><<<dim3(8, 32), 256, 0, stream>>>(
        NX, WoT, bo, x, X1, NTOK, DMODEL, DMODEL);

    // LN2
    stl_layernorm_bf16<<<NTOK, 256, 0, stream>>>(X1, g2, b2, NX);

    // FFN1 (+ exact GELU) -> Hb (bf16)
    stl_gemm_wmma<OUT_BF16,  true><<<dim3(32, 32), 256, 0, stream>>>(
        NX, W1T, bf1, nullptr, Hb, NTOK, DFF, DMODEL);

    // FFN2 + residual -> d_out (f32)
    stl_gemm_wmma<OUT_F32RES, false><<<dim3(8, 32), 256, 0, stream>>>(
        Hb, W2T, bf2, X1, (float*)d_out, NTOK, DMODEL, DFF);
}
